// Decoder_31834297598272
// MI455X (gfx1250) — matrix operand order repaired and checked by reading, not compiled, not run on hardware
//
#include <hip/hip_runtime.h>
#include <hip/hip_bf16.h>
#include <math.h>

// ---------------------------------------------------------------------------
// Tacotron2 decoder for MI455X (gfx1250): bf16 WMMA GEMMs (fp32 accum),
// persistent-in-L2 weights, per-step kernel chain launched directly on the stream.
// ---------------------------------------------------------------------------

typedef __attribute__((ext_vector_type(16))) __bf16 bf16x16;
typedef __attribute__((ext_vector_type(8)))  __bf16 bf16x8;
typedef __attribute__((ext_vector_type(8)))  float  f32x8;

#define N_MELC  80
#define PRENETD 256
#define EDIM    512
#define RNN     1024
#define ATTD    128
#define LOCF    32
#define LOCK    31
#define BB      32
#define TIN     512
#define TOUT    256

// ---- WMMA helper (CDNA5: D = A(16x32 bf16) x B(32x16 bf16) + C(f32)) ------
__device__ __forceinline__ f32x8 wmma_bf16(bf16x16 a, bf16x16 b, f32x8 c) {
  return __builtin_amdgcn_wmma_f32_16x16x32_bf16(
      /*neg_a=*/false, a, /*neg_b=*/false, b,
      /*c_mod=*/(short)0, c, /*reuse_a=*/false, /*reuse_b=*/false);
}

// A-matrix 16x32 bf16 per-lane layout (ISA 7.12.2):
// lane L: M=L%16; elems 0..7 -> K=(L/16)*8+e ; elems 8..15 -> K=16+(L/16)*8+(e-8)
__device__ __forceinline__ bf16x16 load_a16(const __bf16* A, int lda, int m0,
                                            int k0, int lane) {
  int row = m0 + (lane & 15);
  int half = lane >> 4;
  const __bf16* p = A + row * lda + k0 + half * 8;
  bf16x8 lo = *(const bf16x8*)p;
  bf16x8 hi = *(const bf16x8*)(p + 16);
  return __builtin_shufflevector(lo, hi, 0,1,2,3,4,5,6,7,
                                         8,9,10,11,12,13,14,15);
}

// B-matrix 32x16: lane L: N=L%16; elems 0..7 -> K=(L/16)*8+e ; elems 8..15 -> K=16+(L/16)*8+(e-8).
// W stored row-major [N,K] (i.e. B = W^T), so each lane reads two runs of 8 contiguous bf16.
__device__ __forceinline__ bf16x16 load_b16(const __bf16* W, int ldb, int n0,
                                            int k0, int lane) {
  int col = n0 + (lane & 15);
  int half = lane >> 4;
  const __bf16* p = W + col * ldb + k0 + half * 8;
  bf16x8 lo = *(const bf16x8*)p;
  bf16x8 hi = *(const bf16x8*)(p + 16);
  return __builtin_shufflevector(lo, hi, 0,1,2,3,4,5,6,7,
                                         8,9,10,11,12,13,14,15);
}

__device__ __forceinline__ float sigmoidf_(float x) {
  return 1.0f / (1.0f + __expf(-x));
}

__device__ __forceinline__ unsigned keep_mask(unsigned seed, unsigned idx) {
  unsigned u = idx ^ (seed * 0x9E3779B9u);
  u *= 0x9E3779B1u; u ^= u >> 16; u *= 0x85EBCA6Bu; u ^= u >> 13;
  return u & 1u;   // deterministic p=0.5 dropout
}

// ---------------------------- setup kernels --------------------------------
__global__ void cvt_kernel(const float* __restrict__ src, __bf16* __restrict__ dst, int n) {
  for (int i = blockIdx.x * blockDim.x + threadIdx.x; i < n;
       i += gridDim.x * blockDim.x)
    dst[i] = (__bf16)src[i];
}

__global__ void cvt_pad_kernel(const float* __restrict__ src, __bf16* __restrict__ dst,
                               int ri, int ci, int ro, int co) {
  int n = ro * co;
  for (int i = blockIdx.x * blockDim.x + threadIdx.x; i < n;
       i += gridDim.x * blockDim.x) {
    int r = i / co, c = i % co;
    dst[i] = (r < ri && c < ci) ? (__bf16)src[r * ci + c] : (__bf16)0.0f;
  }
}

// X0[t*32+b][c] = (t==0 || c>=80) ? 0 : decoder_inputs[b][c][t-1]   ([8192,96])
__global__ void buildX0_kernel(const float* __restrict__ dec, __bf16* __restrict__ X0) {
  int n = TOUT * BB * 96;
  for (int i = blockIdx.x * blockDim.x + threadIdx.x; i < n;
       i += gridDim.x * blockDim.x) {
    int row = i / 96, c = i % 96;
    int t = row >> 5, b = row & 31;
    float v = (t == 0 || c >= N_MELC) ? 0.0f
                                      : dec[b * N_MELC * TOUT + c * TOUT + (t - 1)];
    X0[i] = (__bf16)v;
  }
}

// ------------------ generic WMMA GEMM: out = A @ W^T -----------------------
// A bf16 [M,K] row-major, W bf16 [N,K] row-major. 8 waves/block, 1 tile/wave.
// mode 0: fp32 store; mode 1: relu + 0.5-dropout(*2) -> bf16 store.
__global__ void gemm_bf16_kernel(const __bf16* __restrict__ A,
                                 const __bf16* __restrict__ W, int K, int N,
                                 float* __restrict__ outf, __bf16* __restrict__ outb,
                                 int mode, unsigned seed) {
  int lane = threadIdx.x & 31, wave = threadIdx.x >> 5;
  int tile = blockIdx.x * 8 + wave;
  int ntiles = N >> 4;
  int mt = tile / ntiles, nt = tile % ntiles;
  f32x8 acc = {};
  for (int k0 = 0; k0 < K; k0 += 32) {
    bf16x16 a = load_a16(A, K, mt * 16, k0, lane);
    bf16x16 b = load_b16(W, K, nt * 16, k0, lane);
    acc = wmma_bf16(a, b, acc);
  }
  int n = nt * 16 + (lane & 15);
  int mbase = mt * 16 + ((lane >> 4) << 3);
  #pragma unroll
  for (int r = 0; r < 8; ++r) {
    int m = mbase + r;
    if (mode == 0) {
      outf[m * N + n] = acc[r];
    } else {
      float v = fmaxf(acc[r], 0.0f);
      v = keep_mask(seed, (unsigned)(m * N + n)) ? 2.0f * v : 0.0f;
      outb[m * N + n] = (__bf16)v;
    }
  }
}

// -------- LSTM gates: gates[32,4096] = concat(A0,A1)@Wih^T + H@Whh^T + b ---
// 32 blocks x 8 waves; each wave: one 16-col tile x both 16-row tiles.
__global__ void gates_kernel(const __bf16* __restrict__ A0, int K0,
                             const __bf16* __restrict__ A1, int K1,
                             const __bf16* __restrict__ Wih,
                             const __bf16* __restrict__ H,
                             const __bf16* __restrict__ Whh,
                             const float* __restrict__ bih,
                             const float* __restrict__ bhh,
                             float* __restrict__ gates) {
  int lane = threadIdx.x & 31, wave = threadIdx.x >> 5;
  int n0 = (blockIdx.x * 8 + wave) * 16;
  int KA = K0 + K1;
  f32x8 c0 = {}, c1 = {};
  for (int k0 = 0; k0 < KA; k0 += 32) {
    bf16x16 a0, a1;
    if (k0 < K0) { a0 = load_a16(A0, K0, 0, k0, lane);
                   a1 = load_a16(A0, K0, 16, k0, lane); }
    else         { a0 = load_a16(A1, K1, 0, k0 - K0, lane);
                   a1 = load_a16(A1, K1, 16, k0 - K0, lane); }
    bf16x16 bt = load_b16(Wih, KA, n0, k0, lane);
    c0 = wmma_bf16(a0, bt, c0);
    c1 = wmma_bf16(a1, bt, c1);
  }
  for (int k0 = 0; k0 < RNN; k0 += 32) {
    bf16x16 a0 = load_a16(H, RNN, 0, k0, lane);
    bf16x16 a1 = load_a16(H, RNN, 16, k0, lane);
    bf16x16 bt = load_b16(Whh, RNN, n0, k0, lane);
    c0 = wmma_bf16(a0, bt, c0);
    c1 = wmma_bf16(a1, bt, c1);
  }
  int n = n0 + (lane & 15);
  float bias = bih[n] + bhh[n];
  int mb = (lane >> 4) << 3;
  #pragma unroll
  for (int r = 0; r < 8; ++r) {
    gates[(mb + r) * 4096 + n]      = c0[r] + bias;
    gates[(16 + mb + r) * 4096 + n] = c1[r] + bias;
  }
}

// ---------------- LSTM elementwise (gate order i,f,g,o) --------------------
__global__ void lstm_kernel(const float* __restrict__ gates, float* __restrict__ c,
                            __bf16* __restrict__ hbf) {
  int idx = blockIdx.x * blockDim.x + threadIdx.x;
  if (idx >= BB * RNN) return;
  int b = idx >> 10, u = idx & 1023;
  const float* g = gates + b * 4096;
  float gi = sigmoidf_(g[u]);
  float gf = sigmoidf_(g[1024 + u]);
  float gg = tanhf(g[2048 + u]);
  float go = sigmoidf_(g[3072 + u]);
  float c2 = gf * c[idx] + gi * gg;
  c[idx] = c2;
  hbf[idx] = (__bf16)(go * tanhf(c2));
}

// --- fused location conv + loc-dense(WMMA) + tanh + v-dot + mask -----------
// grid = B*32 blocks (one block = one b, 16 t-rows), 256 threads = 8 waves.
__global__ void energies_kernel(const float* __restrict__ aw,
                                const float* __restrict__ awc,
                                const float* __restrict__ convW,   // [32,2,31] f32
                                const __bf16* __restrict__ locD,   // [128,32] bf16
                                const float* __restrict__ pq,      // [32,128]
                                const float* __restrict__ pm,      // [32*512,128]
                                const float* __restrict__ vW,      // [128]
                                const int* __restrict__ memlen,
                                float* __restrict__ energies) {
  int b = blockIdx.x >> 5;
  int t0 = (blockIdx.x & 31) * 16;
  int tid = threadIdx.x;
  int lane = tid & 31, wave = tid >> 5;
  __shared__ float shw[2][48];
  __shared__ float shloc[16][32];
  __shared__ float sher[16];
  if (tid < 92) {
    int ch = tid / 46, ix = tid % 46;
    int pos = t0 - 15 + ix;
    const float* src = ch ? awc : aw;
    shw[ch][ix] = (pos >= 0 && pos < TIN) ? src[b * TIN + pos] : 0.0f;
  }
  if (tid >= 92 && tid < 108) sher[tid - 92] = 0.0f;
  __syncthreads();
  // location conv: loc[m][f] over 2 channels x 31 taps ("same" padding)
  for (int idx = tid; idx < 16 * 32; idx += blockDim.x) {
    int m = idx >> 5, f = idx & 31;
    float s = 0.0f;
    #pragma unroll
    for (int c = 0; c < 2; ++c) {
      const float* wp = convW + (f * 2 + c) * LOCK;
      for (int k = 0; k < LOCK; ++k) s += shw[c][m + k] * wp[k];
    }
    shloc[m][f] = s;
  }
  __syncthreads();
  // WMMA: loc[16x32] @ locD^T -> [16 x 128]; wave w owns 16 dims.
  bf16x16 a;
  {
    int m = lane & 15, half = lane >> 4;
    #pragma unroll
    for (int e = 0; e < 8; ++e) a[e] = (__bf16)shloc[m][half * 8 + e];
    #pragma unroll
    for (int e = 0; e < 8; ++e) a[e + 8] = (__bf16)shloc[m][16 + half * 8 + e];
  }
  bf16x16 bt = load_b16(locD, 32, wave * 16, 0, lane);
  f32x8 acc = {};
  acc = wmma_bf16(a, bt, acc);
  int n = wave * 16 + (lane & 15);
  int half = lane >> 4;
  float pqv = pq[b * ATTD + n];
  float vwn = vW[n];
  #pragma unroll
  for (int r = 0; r < 8; ++r) {
    int m = r + (half << 3);
    float e = tanhf(acc[r] + pqv + pm[(b * TIN + t0 + m) * ATTD + n]) * vwn;
    for (int off = 1; off < 16; off <<= 1) e += __shfl_xor(e, off, 32);
    if ((lane & 15) == 0) atomicAdd(&sher[m], e);
  }
  __syncthreads();
  if (tid < 16) {
    int t = t0 + tid;
    energies[b * TIN + t] = (t >= memlen[b]) ? -1e30f : sher[tid];
  }
}

// ------------- softmax over T_IN + cum-weights + context + alignments ------
__global__ void softmax_ctx_kernel(const float* __restrict__ energies,
                                   float* __restrict__ aw, float* __restrict__ awc,
                                   const float* __restrict__ memF,
                                   __bf16* __restrict__ ctxb,
                                   float* __restrict__ alignOut, int step) {
  int b = blockIdx.x, tid = threadIdx.x;
  __shared__ float sh[TIN];
  float e = energies[b * TIN + tid];
  sh[tid] = e; __syncthreads();
  for (int s = 256; s > 0; s >>= 1) {
    if (tid < s) sh[tid] = fmaxf(sh[tid], sh[tid + s]);
    __syncthreads();
  }
  float mx = sh[0]; __syncthreads();
  float ex = __expf(e - mx);
  sh[tid] = ex; __syncthreads();
  for (int s = 256; s > 0; s >>= 1) {
    if (tid < s) sh[tid] += sh[tid + s];
    __syncthreads();
  }
  float total = sh[0]; __syncthreads();
  float w = ex / total;
  aw[b * TIN + tid] = w;
  awc[b * TIN + tid] += w;
  alignOut[(b * TOUT + step) * TIN + tid] = w;
  sh[tid] = w; __syncthreads();
  // ctx[b,e] = sum_t aw[t] * memory[b,t,e]  (thread tid owns dim e=tid)
  float accv = 0.0f;
  const float* mrow = memF + (size_t)b * TIN * EDIM + tid;
  for (int t = 0; t < TIN; ++t) accv += sh[t] * mrow[t * EDIM];
  ctxb[b * EDIM + tid] = (__bf16)accv;
}

// ---------- projection (WMMA) + stop-gate; one block, 16 waves -------------
__global__ void proj_kernel(const __bf16* __restrict__ Hb,   // dh [32,1024]
                            const __bf16* __restrict__ Cb,   // ctx [32,512]
                            const __bf16* __restrict__ Wp,   // [96,1536] padded
                            const float* __restrict__ projB,
                            const float* __restrict__ gateW,
                            const float* __restrict__ gateB,
                            float* __restrict__ melOut, float* __restrict__ gateOut,
                            int step) {
  int tid = threadIdx.x;
  int lane = tid & 31, wave = tid >> 5;
  if (wave < 12) {                       // 2 m-tiles x 6 n-tiles
    int mt = wave / 6, nt = wave % 6;
    f32x8 acc = {};
    for (int k0 = 0; k0 < 1536; k0 += 32) {
      bf16x16 a = (k0 < 1024) ? load_a16(Hb, 1024, mt * 16, k0, lane)
                              : load_a16(Cb, 512, mt * 16, k0 - 1024, lane);
      bf16x16 bt = load_b16(Wp, 1536, nt * 16, k0, lane);
      acc = wmma_bf16(a, bt, acc);
    }
    int n = nt * 16 + (lane & 15);
    if (n < N_MELC) {
      float bias = projB[n];
      int mb = mt * 16 + ((lane >> 4) << 3);
      #pragma unroll
      for (int r = 0; r < 8; ++r)
        melOut[((mb + r) * TOUT + step) * N_MELC + n] = acc[r] + bias;
    }
  }
  if (tid < BB) {
    float g = gateB[0];
    const __bf16* h = Hb + tid * 1024;
    const __bf16* cx = Cb + tid * 512;
    for (int j = 0; j < 1024; ++j) g += (float)h[j] * gateW[j];
    for (int j = 0; j < 512; ++j)  g += (float)cx[j] * gateW[1024 + j];
    gateOut[tid * TOUT + step] = g;
  }
}

// ---------------------------------------------------------------------------
extern "C" void kernel_launch(void* const* d_in, const int* in_sizes, int n_in,
                              void* d_out, int out_size, void* d_ws, size_t ws_size,
                              hipStream_t stream) {
  // Input order: memory, decoder_inputs, memory_lengths, then params dict order.
  const float* memF     = (const float*)d_in[0];
  const float* decInp   = (const float*)d_in[1];
  const int*   memLen   = (const int*)d_in[2];
  const float* pW1      = (const float*)d_in[3];
  const float* pW2      = (const float*)d_in[4];
  const float* attWih   = (const float*)d_in[5];
  const float* attWhh   = (const float*)d_in[6];
  const float* attBih   = (const float*)d_in[7];
  const float* attBhh   = (const float*)d_in[8];
  const float* qW       = (const float*)d_in[9];
  const float* memW     = (const float*)d_in[10];
  const float* vW       = (const float*)d_in[11];
  const float* locConvW = (const float*)d_in[12];
  const float* locDenseW= (const float*)d_in[13];
  const float* decWih   = (const float*)d_in[14];
  const float* decWhh   = (const float*)d_in[15];
  const float* decBih   = (const float*)d_in[16];
  const float* decBhh   = (const float*)d_in[17];
  const float* projW    = (const float*)d_in[18];
  const float* projB    = (const float*)d_in[19];
  const float* gateW    = (const float*)d_in[20];
  const float* gateB    = (const float*)d_in[21];
  (void)in_sizes; (void)n_in; (void)out_size; (void)ws_size;

  float* out      = (float*)d_out;
  float* melOut   = out;                          // [32,256,80]
  float* gateOut  = out + BB * TOUT * N_MELC;     // [32,256]
  float* alignOut = gateOut + BB * TOUT;          // [32,256,512]

  char* base = (char*)d_ws;
  size_t off = 0;
  auto carve = [&](size_t bytes) -> char* {
    char* p = base + off;
    off = (off + bytes + 255) & ~(size_t)255;
    return p;
  };
  // bf16 weight/activation buffers (all stay resident in the 192MB L2)
  __bf16* wAttWih = (__bf16*)carve((size_t)4096 * 768 * 2);
  __bf16* wAttWhh = (__bf16*)carve((size_t)4096 * 1024 * 2);
  __bf16* wDecWih = (__bf16*)carve((size_t)4096 * 1536 * 2);
  __bf16* wDecWhh = (__bf16*)carve((size_t)4096 * 1024 * 2);
  __bf16* wQ      = (__bf16*)carve((size_t)128 * 1024 * 2);
  __bf16* wMem    = (__bf16*)carve((size_t)128 * 512 * 2);
  __bf16* wPre1   = (__bf16*)carve((size_t)256 * 96 * 2);   // padded K 80->96
  __bf16* wPre2   = (__bf16*)carve((size_t)256 * 256 * 2);
  __bf16* wLocD   = (__bf16*)carve((size_t)128 * 32 * 2);
  __bf16* wProj   = (__bf16*)carve((size_t)96 * 1536 * 2);  // padded N 80->96
  __bf16* memB    = (__bf16*)carve((size_t)BB * TIN * EDIM * 2);
  __bf16* X0      = (__bf16*)carve((size_t)TOUT * BB * 96 * 2);
  __bf16* H1      = (__bf16*)carve((size_t)TOUT * BB * 256 * 2);
  __bf16* pren    = (__bf16*)carve((size_t)TOUT * BB * 256 * 2);
  float*  pmF     = (float*)carve((size_t)BB * TIN * ATTD * 4);
  float*  gatesWs = (float*)carve((size_t)BB * 4096 * 4);
  float*  pqF     = (float*)carve((size_t)BB * ATTD * 4);
  float*  enF     = (float*)carve((size_t)BB * TIN * 4);
  float*  awF     = (float*)carve((size_t)BB * TIN * 4);
  float*  awcF    = (float*)carve((size_t)BB * TIN * 4);
  float*  acF     = (float*)carve((size_t)BB * RNN * 4);
  float*  dcF     = (float*)carve((size_t)BB * RNN * 4);
  __bf16* ahB     = (__bf16*)carve((size_t)BB * RNN * 2);
  __bf16* dhB     = (__bf16*)carve((size_t)BB * RNN * 2);
  __bf16* ctxB    = (__bf16*)carve((size_t)BB * EDIM * 2);

  // ---- zero recurrent state (ws is poisoned; must re-init every call) ----
  hipMemsetAsync(awF,  0, (size_t)BB * TIN * 4, stream);
  hipMemsetAsync(awcF, 0, (size_t)BB * TIN * 4, stream);
  hipMemsetAsync(acF,  0, (size_t)BB * RNN * 4, stream);
  hipMemsetAsync(dcF,  0, (size_t)BB * RNN * 4, stream);
  hipMemsetAsync(ahB,  0, (size_t)BB * RNN * 2, stream);
  hipMemsetAsync(dhB,  0, (size_t)BB * RNN * 2, stream);
  hipMemsetAsync(ctxB, 0, (size_t)BB * EDIM * 2, stream);

  // ---- weight / input conversion to bf16 ----
  auto cvt = [&](const float* s, __bf16* d, int n) {
    int blocks = (n + 1023) / 1024; if (blocks > 2048) blocks = 2048;
    cvt_kernel<<<blocks, 256, 0, stream>>>(s, d, n);
  };
  cvt(attWih, wAttWih, 4096 * 768);
  cvt(attWhh, wAttWhh, 4096 * 1024);
  cvt(decWih, wDecWih, 4096 * 1536);
  cvt(decWhh, wDecWhh, 4096 * 1024);
  cvt(qW, wQ, 128 * 1024);
  cvt(memW, wMem, 128 * 512);
  cvt(pW2, wPre2, 256 * 256);
  cvt(locDenseW, wLocD, 128 * 32);
  cvt(memF, memB, BB * TIN * EDIM);
  cvt_pad_kernel<<<64, 256, 0, stream>>>(pW1, wPre1, 256, 80, 256, 96);
  cvt_pad_kernel<<<576, 256, 0, stream>>>(projW, wProj, 80, 1536, 96, 1536);
  buildX0_kernel<<<768, 256, 0, stream>>>(decInp, X0);

  // ---- prenet for all 256 steps (teacher-forced) ----
  gemm_bf16_kernel<<<1024, 256, 0, stream>>>(X0, wPre1, 96, 256,
                                             nullptr, H1, 1, 0x1234u);
  gemm_bf16_kernel<<<1024, 256, 0, stream>>>(H1, wPre2, 256, 256,
                                             nullptr, pren, 1, 0x5678u);
  // ---- processed_memory = memory @ mem_W^T  ([16384,512]x[512,128]) ----
  gemm_bf16_kernel<<<1024, 256, 0, stream>>>(memB, wMem, 512, 128,
                                             pmF, nullptr, 0, 0);

  // ---- 256 sequential decode steps ----
  for (int step = 0; step < TOUT; ++step) {
    const __bf16* prenT = pren + (size_t)step * BB * 256;
    // attention LSTM gates + cell
    gates_kernel<<<32, 256, 0, stream>>>(prenT, 256, ctxB, 512,
                                         wAttWih, ahB, wAttWhh,
                                         attBih, attBhh, gatesWs);
    lstm_kernel<<<128, 256, 0, stream>>>(gatesWs, acF, ahB);
    // query projection pq = ah @ q_W^T  ([32,1024]x[1024,128])
    gemm_bf16_kernel<<<2, 256, 0, stream>>>(ahB, wQ, 1024, 128,
                                            pqF, nullptr, 0, 0);
    // location-sensitive attention energies
    energies_kernel<<<BB * 32, 256, 0, stream>>>(awF, awcF, locConvW, wLocD,
                                                 pqF, pmF, vW, memLen, enF);
    // softmax + cumulative weights + context + alignment output
    softmax_ctx_kernel<<<BB, 512, 0, stream>>>(enF, awF, awcF, memF, ctxB,
                                               alignOut, step);
    // decoder LSTM gates + cell
    gates_kernel<<<32, 256, 0, stream>>>(ahB, 1024, ctxB, 512,
                                         wDecWih, dhB, wDecWhh,
                                         decBih, decBhh, gatesWs);
    lstm_kernel<<<128, 256, 0, stream>>>(gatesWs, dcF, dhB);
    // mel projection + stop gate
    proj_kernel<<<1, 512, 0, stream>>>(dhB, ctxB, wProj, projB,
                                       gateW, gateB, melOut, gateOut, step);
  }
}
